// MultiHeadAttention_4982162063738
// MI455X (gfx1250) — matrix operand order repaired and checked by reading, not compiled, not run on hardware
//
#include <hip/hip_runtime.h>
#include <hip/hip_bf16.h>

#define BB 4
#define SS 2048
#define EE 1024
#define HH 16
#define DKK 64

typedef __attribute__((ext_vector_type(16))) __bf16 v16bf;
typedef __attribute__((ext_vector_type(8)))  __bf16 v8bf;
typedef __attribute__((ext_vector_type(8)))  float  v8f;
typedef __attribute__((ext_vector_type(4)))  float  v4f;

static __device__ __forceinline__ v8f wmma_bf16(v16bf a, v16bf b, v8f c) {
  return __builtin_amdgcn_wmma_f32_16x16x32_bf16(
      /*neg_a=*/false, a, /*neg_b=*/false, b,
      /*c_mod=*/(short)0, c, /*reuse_a=*/false, /*reuse_b=*/false);
}

static __device__ __forceinline__ v16bf make_frag(v8bf lo, v8bf hi) {
  v16bf r;
#pragma unroll
  for (int i = 0; i < 8; ++i) { r[i] = lo[i]; r[i + 8] = hi[i]; }
  return r;
}

// A-fragment (16x32, 16-bit): lane holds row M=lane%16, K = base..base+7 and
// base+16..base+23 where base = 8*(lane/16).  p must already point at k=base.
static __device__ __forceinline__ v16bf load_a_f32(const float* p) {
  v4f a0 = *(const v4f*)(p);
  v4f a1 = *(const v4f*)(p + 4);
  v4f a2 = *(const v4f*)(p + 16);
  v4f a3 = *(const v4f*)(p + 20);
  v16bf r;
#pragma unroll
  for (int i = 0; i < 4; ++i) {
    r[i]      = (__bf16)a0[i];
    r[i + 4]  = (__bf16)a1[i];
    r[i + 8]  = (__bf16)a2[i];
    r[i + 12] = (__bf16)a3[i];
  }
  return r;
}

static __device__ __forceinline__ v16bf load_a_bf16(const __bf16* p) {
  return make_frag(*(const v8bf*)p, *(const v8bf*)(p + 16));
}

// B-fragment read from a pitched LDS tile: K = base..base+7, base+16..base+23.
static __device__ __forceinline__ v16bf load_b_lds(const __bf16* p) {
  return make_frag(*(const v8bf*)p, *(const v8bf*)(p + 16));
}

// Issue one async global->LDS 16-byte copy (ASYNCcnt-tracked).
static __device__ __forceinline__ void async_b128(unsigned lofs, const void* g) {
  asm volatile("global_load_async_to_lds_b128 %0, %1, off"
               :: "v"(lofs), "v"((unsigned long long)g) : "memory");
}
static __device__ __forceinline__ void wait_async0() {
  asm volatile("s_wait_asynccnt 0" ::: "memory");
}

// ---------------------------------------------------------------------------
// Kernel 0: one-pass f32 -> bf16 weight conversion (8 elems / thread).
// ---------------------------------------------------------------------------
__global__ __launch_bounds__(256) void wcvt_kernel(const float* __restrict__ src,
                                                   __bf16* __restrict__ dst) {
  const size_t i = ((size_t)blockIdx.x * 256 + threadIdx.x) * 8;
  v4f a = *(const v4f*)(src + i);
  v4f b = *(const v4f*)(src + i + 4);
  v8bf o;
#pragma unroll
  for (int j = 0; j < 4; ++j) { o[j] = (__bf16)a[j]; o[j + 4] = (__bf16)b[j]; }
  *(v8bf*)(dst + i) = o;
}

// Pitches chosen so the 16 lanes of a B-fragment read hit 16 distinct banks.
#define WT_PITCH 40   // 64 rows x 32 k tile (weights, V):  row stride 80 B
#define KT_PITCH 72   // 32 rows x 64 d tile (K):           row stride 144 B

// Issue-only staging of a 64(n) x 32(k) bf16 tile: 128 thr x 2 async b128.
static __device__ __forceinline__ void stage_64x32(
    const __bf16* __restrict__ src, size_t row_stride, int row0, int k0,
    __bf16* buf, int tid) {
#pragma unroll
  for (int c = 0; c < 2; ++c) {
    const int chunk = tid * 2 + c;          // 0..255
    const int row = chunk >> 2;             // 0..63
    const int k8 = (chunk & 3) * 8;         // 0,8,16,24
    async_b128((unsigned)(size_t)(buf + row * WT_PITCH + k8),
               src + (size_t)(row0 + row) * row_stride + k0 + k8);
  }
}

// Issue-only staging of a 32(key) x 64(d) K tile: 128 thr x 2 async b128.
static __device__ __forceinline__ void stage_ktile(
    const __bf16* __restrict__ src, size_t base, int kb, __bf16* buf, int tid) {
#pragma unroll
  for (int c = 0; c < 2; ++c) {
    const int chunk = tid * 2 + c;          // 0..255
    const int row = chunk >> 3;             // 0..31  (key)
    const int d8 = (chunk & 7) * 8;         // 0..56
    async_b128((unsigned)(size_t)(buf + row * KT_PITCH + d8),
               src + (base + kb + row) * DKK + d8);
  }
}

// ---------------------------------------------------------------------------
// Kernel 1: input projections.  C[m, f] = sum_e A[m,e] * W[f,e]  (W in bf16)
// mode 0: q, scaled by 1/sqrt(DK), layout [B,H,S,DK]
// mode 1: k,                        layout [B,H,S,DK]
// mode 2: v,                        layout [B,H,DK,S]  (transposed for P@V)
// Wave tile 16(M)x64(N); 4 waves stacked in M; weight tiles double-buffered
// in LDS via async global->LDS loads overlapped with WMMA.
// ---------------------------------------------------------------------------
__global__ __launch_bounds__(128) void proj_kernel(
    const float* __restrict__ A, const __bf16* __restrict__ Wb,
    __bf16* __restrict__ out, int mode) {
  __shared__ __align__(16) __bf16 wtile[2][64 * WT_PITCH];

  const int tid = threadIdx.x;
  const int wave = tid >> 5;
  const int lane = tid & 31;
  const int half = lane >> 4, lr = lane & 15;
  const int m0 = blockIdx.x * 64 + wave * 16;
  const int n0 = blockIdx.y * 64;

  const float* arow = A + (size_t)(m0 + lr) * EE;
  v8f acc[4] = {};

  stage_64x32(Wb, EE, n0, 0, wtile[0], tid);       // prologue
  for (int it = 0; it < EE / 32; ++it) {
    const int k = it * 32;
    const __bf16* cur = wtile[it & 1];
    wait_async0();                                  // own loads done
    __syncthreads();                                // tile resident, prev reads done
    if (it + 1 < EE / 32)
      stage_64x32(Wb, EE, n0, k + 32, wtile[(it + 1) & 1], tid);
    v16bf af = load_a_f32(arow + k + half * 8);
#pragma unroll
    for (int t = 0; t < 4; ++t) {
      const __bf16* wl = cur + (t * 16 + lr) * WT_PITCH + half * 8;
      acc[t] = wmma_bf16(af, load_b_lds(wl), acc[t]);
    }
  }
  const float scale = (mode == 0) ? 0.125f : 1.0f;  // 1/sqrt(64) folded into q
#pragma unroll
  for (int t = 0; t < 4; ++t) {
    const int f = n0 + t * 16 + lr;
    const int h = f >> 6, d = f & 63;
#pragma unroll
    for (int r = 0; r < 8; ++r) {
      const int mrow = m0 + r + half * 8;       // flat token index (b*S + s)
      const int b = mrow >> 11, s = mrow & (SS - 1);
      const float v = acc[t][r] * scale;
      size_t idx;
      if (mode == 2) idx = (((size_t)(b * HH + h)) * DKK + d) * SS + s;
      else           idx = (((size_t)(b * HH + h)) * SS + s) * DKK + d;
      out[idx] = (__bf16)v;
    }
  }
}

// ---------------------------------------------------------------------------
// Kernel 2: flash attention.  One block owns 64 queries of one (b,h); the
// 32-key K/V tiles are shared by all 4 waves: staged once per block into
// double-buffered LDS via async loads, overlapped with the WMMA stream.
// ---------------------------------------------------------------------------
__global__ __launch_bounds__(128) void attn_kernel(
    const __bf16* __restrict__ q, const __bf16* __restrict__ k,
    const __bf16* __restrict__ vt, const unsigned char* __restrict__ mask,
    __bf16* __restrict__ out) {
  __shared__ __align__(16) __bf16 ktile[2][32 * KT_PITCH];  // [key][d]
  __shared__ __align__(16) __bf16 vtile[2][64 * WT_PITCH];  // [d][key]
  __shared__ __align__(16) __bf16 plds[4][16 * 32];         // per-wave P

  const int tid = threadIdx.x;
  const int wave = tid >> 5;
  const int lane = tid & 31;
  const int half = lane >> 4, lr = lane & 15;
  const int bh = blockIdx.y;
  const int b = bh >> 4, h = bh & 15;
  const int q0 = blockIdx.x * 64 + wave * 16;

  const size_t bhS = (size_t)bh * SS;

  // Q A-fragments covering DK=64 (two 16x32 tiles)
  const __bf16* qrow = q + (bhS + q0 + lr) * DKK;
  const v16bf qf0 = load_a_bf16(qrow + half * 8);
  const v16bf qf1 = load_a_bf16(qrow + 32 + half * 8);

  float mrow[8], lrow[8];
#pragma unroll
  for (int r = 0; r < 8; ++r) { mrow[r] = -3.0e38f; lrow[r] = 0.f; }
  v8f oacc[4] = {};

  const unsigned char* mp = mask + (size_t)b * SS;
  const __bf16* vbase = vt + (size_t)bh * DKK * SS;

  stage_ktile(k, bhS, 0, ktile[0], tid);            // prologue
  stage_64x32(vbase, SS, 0, 0, vtile[0], tid);      // V: rows=d, cols=key
  for (int it = 0; it < SS / 32; ++it) {
    const int kb = it * 32;
    const __bf16* kc = ktile[it & 1];
    const __bf16* vc = vtile[it & 1];
    wait_async0();
    __syncthreads();
    if (it + 1 < SS / 32) {
      stage_ktile(k, bhS, kb + 32, ktile[(it + 1) & 1], tid);
      stage_64x32(vbase, SS, 0, kb + 32, vtile[(it + 1) & 1], tid);
    }
    // ---- scores S = q @ k^T for two 16-key subtiles ----
    v8f sc[2];
#pragma unroll
    for (int kt = 0; kt < 2; ++kt) {
      const __bf16* kl = kc + (kt * 16 + lr) * KT_PITCH + half * 8;
      v8f s = {};
      s = wmma_bf16(qf0, load_b_lds(kl), s);        // d = 0..31
      s = wmma_bf16(qf1, load_b_lds(kl + 32), s);   // d = 32..63
      if (mp[kb + kt * 16 + lr]) {  // key mask: this lane IS one key column
#pragma unroll
        for (int r = 0; r < 8; ++r) s[r] = -1.0e9f;
      }
      sc[kt] = s;
    }
    // ---- online softmax (row = 16 lanes of one accumulator slot) ----
#pragma unroll
    for (int r = 0; r < 8; ++r) {
      float x = fmaxf(sc[0][r], sc[1][r]);
#pragma unroll
      for (int mo = 1; mo < 16; mo <<= 1) x = fmaxf(x, __shfl_xor(x, mo, 32));
      const float mn = fmaxf(mrow[r], x);
      const float corr = __expf(mrow[r] - mn);
      mrow[r] = mn;
      const float p0 = __expf(sc[0][r] - mn);
      const float p1 = __expf(sc[1][r] - mn);
      sc[0][r] = p0; sc[1][r] = p1;
      float sum = p0 + p1;
#pragma unroll
      for (int mo = 1; mo < 16; mo <<= 1) sum += __shfl_xor(sum, mo, 32);
      lrow[r] = lrow[r] * corr + sum;
#pragma unroll
      for (int t = 0; t < 4; ++t) oacc[t][r] *= corr;
    }
    // ---- re-stripe P from C-layout to A-fragment layout via LDS ----
#pragma unroll
    for (int kt = 0; kt < 2; ++kt)
#pragma unroll
      for (int r = 0; r < 8; ++r)
        plds[wave][(r + half * 8) * 32 + kt * 16 + lr] = (__bf16)sc[kt][r];
    asm volatile("s_wait_dscnt 0" ::: "memory");  // in-wave LDS WAR/RAW fence
    const v16bf pf =
        make_frag(*(const v8bf*)&plds[wave][lr * 32 + half * 8],
                  *(const v8bf*)&plds[wave][lr * 32 + half * 8 + 16]);
    // ---- O += P @ V ----
#pragma unroll
    for (int t = 0; t < 4; ++t) {
      const __bf16* vl = vc + (t * 16 + lr) * WT_PITCH + half * 8;
      oacc[t] = wmma_bf16(pf, load_b_lds(vl), oacc[t]);
    }
  }
  // ---- normalize and store attn output, heads re-concatenated [B,S,E] ----
#pragma unroll
  for (int t = 0; t < 4; ++t) {
#pragma unroll
    for (int r = 0; r < 8; ++r) {
      const int s = q0 + r + half * 8;
      const float v = oacc[t][r] / lrow[r];
      out[((size_t)b * SS + s) * EE + h * 64 + t * 16 + lr] = (__bf16)v;
    }
  }
}

// ---------------------------------------------------------------------------
// Kernel 3: output projection + residual.  out = res + attn @ W_O^T  (f32)
// Same double-buffered LDS weight staging; A is bf16.
// ---------------------------------------------------------------------------
__global__ __launch_bounds__(128) void oproj_kernel(
    const __bf16* __restrict__ A, const __bf16* __restrict__ Wb,
    const float* __restrict__ res, float* __restrict__ out) {
  __shared__ __align__(16) __bf16 wtile[2][64 * WT_PITCH];

  const int tid = threadIdx.x;
  const int wave = tid >> 5;
  const int lane = tid & 31;
  const int half = lane >> 4, lr = lane & 15;
  const int m0 = blockIdx.x * 64 + wave * 16;
  const int n0 = blockIdx.y * 64;

  const __bf16* arow = A + (size_t)(m0 + lr) * EE;
  v8f acc[4] = {};

  stage_64x32(Wb, EE, n0, 0, wtile[0], tid);
  for (int it = 0; it < EE / 32; ++it) {
    const int k = it * 32;
    const __bf16* cur = wtile[it & 1];
    wait_async0();
    __syncthreads();
    if (it + 1 < EE / 32)
      stage_64x32(Wb, EE, n0, k + 32, wtile[(it + 1) & 1], tid);
    v16bf af = load_a_bf16(arow + k + half * 8);
#pragma unroll
    for (int t = 0; t < 4; ++t) {
      const __bf16* wl = cur + (t * 16 + lr) * WT_PITCH + half * 8;
      acc[t] = wmma_bf16(af, load_b_lds(wl), acc[t]);
    }
  }
#pragma unroll
  for (int t = 0; t < 4; ++t) {
#pragma unroll
    for (int r = 0; r < 8; ++r) {
      const size_t idx =
          (size_t)(m0 + r + half * 8) * EE + n0 + t * 16 + lr;
      out[idx] = res[idx] + acc[t][r];
    }
  }
}

extern "C" void kernel_launch(void* const* d_in, const int* in_sizes, int n_in,
                              void* d_out, int out_size, void* d_ws,
                              size_t ws_size, hipStream_t stream) {
  (void)in_sizes; (void)n_in; (void)out_size; (void)ws_size;
  const float* Q  = (const float*)d_in[0];
  const float* K  = (const float*)d_in[1];
  const float* V  = (const float*)d_in[2];
  const float* WQ = (const float*)d_in[3];
  const float* WK = (const float*)d_in[4];
  const float* WV = (const float*)d_in[5];
  const float* WO = (const float*)d_in[6];
  const unsigned char* mask = (const unsigned char*)d_in[7];
  float* out = (float*)d_out;

  const size_t n = (size_t)BB * SS * EE;  // 8.4M elems per activation tensor
  const size_t m = (size_t)EE * EE;       // 1.05M elems per weight matrix
  __bf16* qs  = (__bf16*)d_ws;            // [B,H,S,DK], pre-scaled
  __bf16* ks  = qs + n;                   // [B,H,S,DK]
  __bf16* vts = ks + n;                   // [B,H,DK,S]
  __bf16* at  = vts + n;                  // [B,S,E]
  __bf16* wqb = at + n;                   // bf16 weights
  __bf16* wkb = wqb + m;
  __bf16* wvb = wkb + m;
  __bf16* wob = wvb + m;

  const dim3 gw(m / (256 * 8));           // 512 blocks
  wcvt_kernel<<<gw, 256, 0, stream>>>(WQ, wqb);
  wcvt_kernel<<<gw, 256, 0, stream>>>(WK, wkb);
  wcvt_kernel<<<gw, 256, 0, stream>>>(WV, wvb);
  wcvt_kernel<<<gw, 256, 0, stream>>>(WO, wob);

  const dim3 gp(BB * SS / 64, EE / 64);   // 128 x 16 blocks
  proj_kernel<<<gp, 128, 0, stream>>>(Q, wqb, qs, 0);
  proj_kernel<<<gp, 128, 0, stream>>>(K, wkb, ks, 1);
  proj_kernel<<<gp, 128, 0, stream>>>(V, wvb, vts, 2);
  attn_kernel<<<dim3(SS / 64, BB * HH), 128, 0, stream>>>(qs, ks, vts, mask, at);
  oproj_kernel<<<gp, 128, 0, stream>>>(at, wob, Q, out);
}
